// RecursiveUnroll_8572754723149
// MI455X (gfx1250) — matrix operand order repaired and checked by reading, not compiled, not run on hardware
//
#include <hip/hip_runtime.h>

#define B_   128
#define T_   256
#define D_   1024
#define ND3  3072
#define KT_  32          // K tiles of 32 (1024/32)
#define NT_  64          // D/16 column tiles

typedef __attribute__((ext_vector_type(16))) __bf16 v16bf;
typedef __attribute__((ext_vector_type(8)))  __bf16 v8bf;
typedef __attribute__((ext_vector_type(8)))  float  v8f;

__device__ __forceinline__ __bf16 f2bf(float f) {
  unsigned u = __builtin_bit_cast(unsigned, f);
  u += 0x7fffu + ((u >> 16) & 1u);            // round-to-nearest-even
  unsigned short s = (unsigned short)(u >> 16);
  return __builtin_bit_cast(__bf16, s);
}

// ---------------------------------------------------------------------------
// Prologue 1: pack fp32 weights into WMMA-B-ready bf16 fragments.
// Fragment (gg, nt, kt) = 512 bf16, lane-major: lane l owns 16 contiguous
// bf16 = W[k][n] with n = gate*D + nt*16 + (l&15),
//                    k = kt*32 + (l>>4)*16 + e   (e = 0..15)
// contiguous k order, the same order mma_step uses for the A fragment.
// gg = s*3 + gate, s: 0 = kernel, 1 = rec_kernel.
// ---------------------------------------------------------------------------
__global__ __launch_bounds__(256) void pack_weights(
    const float* __restrict__ ker, const float* __restrict__ rec,
    __bf16* __restrict__ Wp) {
  int idx = blockIdx.x * 256 + threadIdx.x;   // < 6*64*32*512
  int pos = idx & 511;
  int l = pos >> 4, e = pos & 15;
  int f = idx >> 9;
  int kt = f & 31; f >>= 5;
  int nt = f & 63; f >>= 6;
  int gg = f;                                 // 0..5
  int s = gg / 3, g = gg % 3;
  int n = g * D_ + nt * 16 + (l & 15);
  int k = kt * 32 + (l >> 4) * 16 + e;
  const float* src = s ? rec : ker;
  Wp[idx] = f2bf(src[k * ND3 + n]);
}

// ---------------------------------------------------------------------------
// Prologue 2: out[:,0,:] = x0 ; h_bf[0] = bf16(x0)
// ---------------------------------------------------------------------------
__global__ __launch_bounds__(256) void init_state(
    const float* __restrict__ x, float* __restrict__ out,
    __bf16* __restrict__ h0) {
  int idx = blockIdx.x * 256 + threadIdx.x;   // < B*D
  int b = idx >> 10, n = idx & 1023;
  float v = x[idx];
  out[b * (T_ * D_) + n] = v;
  h0[idx] = f2bf(v);
}

// ---------------------------------------------------------------------------
// Async staging of one 6 KB K-tile of packed weights into LDS.
// Waves 0..5 (tid < 192, wave-uniform) each own one gate fragment; every
// lane issues two GLOBAL_LOAD_ASYNC_TO_LDS_B128 (IOFFSET applies to both
// the LDS and global address). Tracked by ASYNCcnt; no VGPR data traffic.
// The per-lane source pointer advances by 512 bf16 (1 KB) per staged K-tile.
// ---------------------------------------------------------------------------
__device__ __forceinline__ void stage_async(const __bf16* src, unsigned ldsoff,
                                            bool stager) {
  if (stager) {
    unsigned long long ga = (unsigned long long)(size_t)src;
    asm volatile(
        "global_load_async_to_lds_b128 %0, %1, off\n\t"
        "global_load_async_to_lds_b128 %0, %1, off offset:16"
        :: "v"(ldsoff), "v"(ga) : "memory");
  }
}

__device__ __forceinline__ void wait_async() {
  // scalar wait; no-op for waves with ASYNCcnt == 0, so run it in all waves
  asm volatile("s_wait_asynccnt 0x0" ::: "memory");
}

template <int FIRST>
__device__ __forceinline__ void mma_step(const __bf16* __restrict__ buf,
                                         const __bf16* __restrict__ arow,
                                         int kt, int half, int lane, v8f* acc) {
  // A fragment: lanes 0-15 get K {k0..k0+15}, lanes 16-31
  // get K {k0+16..k0+31} (contiguous, matching packed B).
  int k1 = kt * 32 + half * 16;
  v8bf alo = *(const v8bf*)(arow + k1);
  v8bf ahi = *(const v8bf*)(arow + k1 + 8);
  v16bf a = __builtin_shufflevector(alo, ahi,
      0, 1, 2, 3, 4, 5, 6, 7, 8, 9, 10, 11, 12, 13, 14, 15);

  // Pull all B fragments into distinct registers first so the ds_loads can
  // be issued as one clause and drained while WMMAs overlap.
  const v16bf* bfrag = (const v16bf*)(buf + lane * 16);  // +32 (v16bf) per gg
  v16bf b0 = bfrag[0];
  v16bf b1 = bfrag[32];
  v16bf b2 = bfrag[64];
  if (!FIRST) {
    v16bf b3 = bfrag[96];
    v16bf b4 = bfrag[128];
    v16bf b5 = bfrag[160];
    acc[0] = __builtin_amdgcn_wmma_f32_16x16x32_bf16(false, a, false, b0,
                                                     (short)0, acc[0], false, false);
    acc[1] = __builtin_amdgcn_wmma_f32_16x16x32_bf16(false, a, false, b1,
                                                     (short)0, acc[1], false, false);
    acc[2] = __builtin_amdgcn_wmma_f32_16x16x32_bf16(false, a, false, b2,
                                                     (short)0, acc[2], false, false);
    acc[3] = __builtin_amdgcn_wmma_f32_16x16x32_bf16(false, a, false, b3,
                                                     (short)0, acc[3], false, false);
    acc[4] = __builtin_amdgcn_wmma_f32_16x16x32_bf16(false, a, false, b4,
                                                     (short)0, acc[4], false, false);
    acc[5] = __builtin_amdgcn_wmma_f32_16x16x32_bf16(false, a, false, b5,
                                                     (short)0, acc[5], false, false);
  } else {   // h == 0 on the first step: recurrent gates stay zero
    acc[0] = __builtin_amdgcn_wmma_f32_16x16x32_bf16(false, a, false, b0,
                                                     (short)0, acc[0], false, false);
    acc[1] = __builtin_amdgcn_wmma_f32_16x16x32_bf16(false, a, false, b1,
                                                     (short)0, acc[1], false, false);
    acc[2] = __builtin_amdgcn_wmma_f32_16x16x32_bf16(false, a, false, b2,
                                                     (short)0, acc[2], false, false);
  }
}

// ---------------------------------------------------------------------------
// One GRU timestep. 64 blocks (16 columns each), 8 waves per block (16 rows
// each). 6 bf16 WMMA accumulations per wave (z/r/h for both weight matrices),
// B fragments staged through double-buffered LDS via async-to-LDS copies,
// gates fused in-register. FIRST (t==1: h==0) is compile-time.
// ---------------------------------------------------------------------------
template <int FIRST>
__global__ __launch_bounds__(256) void gru_step(
    const __bf16* __restrict__ Wp, const __bf16* __restrict__ hprev_bf,
    const float* __restrict__ bias, float* __restrict__ out,
    __bf16* __restrict__ hnext_bf, int t) {
  __shared__ __bf16 smem[2][6 * 512];         // 12 KB double buffer
  const int tid  = threadIdx.x;
  const int wave = tid >> 5, lane = tid & 31;
  const int nt   = blockIdx.x;
  const int half = lane >> 4;
  const bool stager = tid < 192;              // wave-uniform (waves 0..5)

  v8f acc[6] = {};

  const int mrow = wave * 16 + (lane & 15);
  const __bf16* arow = hprev_bf + mrow * D_;

  // Per-lane staging state: moving global src pointer + two LDS offsets.
  const int gg = tid >> 5, ll = tid & 31;
  const __bf16* sptr = Wp + (gg * NT_ + nt) * (KT_ * 512) + ll * 16;
  const unsigned lds0 = (unsigned)(size_t)
      (__attribute__((address_space(3))) __bf16*)(&smem[0][0] + gg * 512 + ll * 16);
  const unsigned lds1 = (unsigned)(size_t)
      (__attribute__((address_space(3))) __bf16*)(&smem[1][0] + gg * 512 + ll * 16);

  stage_async(sptr, lds0, stager);  sptr += 512;
  wait_async();
  __syncthreads();

  // Unrolled by 2 so the ping-pong buffer index is a compile-time constant.
  for (int kt = 0; kt < KT_; kt += 2) {
    stage_async(sptr, lds1, stager);  sptr += 512;       // kt+1 < 32 always
    mma_step<FIRST>(smem[0], arow, kt, half, lane, acc);
    wait_async();
    __syncthreads();

    if (kt + 2 < KT_) stage_async(sptr, lds0, stager);
    sptr += 512;
    mma_step<FIRST>(smem[1], arow, kt + 1, half, lane, acc);
    wait_async();
    __syncthreads();
  }

  // Fused gate math. C-tile layout: VGPR i -> row (wave*16 + i + 8*half),
  // col nt*16 + (lane&15).
  const int n = nt * 16 + (lane & 15);
  const float bz = bias[n], br = bias[D_ + n], bh = bias[2 * D_ + n];
#pragma unroll
  for (int i = 0; i < 8; ++i) {
    int m = wave * 16 + i + half * 8;
    float xz = acc[0][i] + bz;
    float xr = acc[1][i] + br;
    float xh = acc[2][i] + bh;
    float z  = 1.0f / (1.0f + __expf(-(xz + acc[3][i])));
    float r  = 1.0f / (1.0f + __expf(-(xr + acc[4][i])));
    float pre = xh + r * acc[5][i];
    float hc  = 2.0f / (1.0f + __expf(-2.0f * pre)) - 1.0f;   // tanh
    float hp  = FIRST ? 0.0f : out[m * (T_ * D_) + (t - 1) * D_ + n];
    float hn  = z * hp + (1.0f - z) * hc;
    out[m * (T_ * D_) + t * D_ + n] = hn;
    hnext_bf[m * D_ + n] = f2bf(hn);
  }
}

// ---------------------------------------------------------------------------
extern "C" void kernel_launch(void* const* d_in, const int* in_sizes, int n_in,
                              void* d_out, int out_size, void* d_ws, size_t ws_size,
                              hipStream_t stream) {
  const float* x    = (const float*)d_in[0];
  const float* ker  = (const float*)d_in[1];
  const float* rec  = (const float*)d_in[2];
  const float* bias = (const float*)d_in[3];
  float* out = (float*)d_out;

  char* ws = (char*)d_ws;
  __bf16* Wp   = (__bf16*)ws;                                  // 12.58 MB packed weights
  __bf16* hbf0 = (__bf16*)(ws + (size_t)6 * 64 * 32 * 512 * 2);
  __bf16* hbf1 = hbf0 + (size_t)B_ * D_;
  __bf16* hb[2] = {hbf0, hbf1};

  pack_weights<<<6 * 64 * 32 * 512 / 256, 256, 0, stream>>>(ker, rec, Wp);
  init_state<<<B_ * D_ / 256, 256, 0, stream>>>(x, out, hbf0);

  gru_step<1><<<NT_, 256, 0, stream>>>(Wp, hb[0], bias, out, hb[1], 1);
  for (int t = 2; t < T_; ++t) {
    gru_step<0><<<NT_, 256, 0, stream>>>(Wp, hb[(t - 1) & 1], bias, out,
                                         hb[t & 1], t);
  }
}
